// RelativePositionalMultiHeadAttention_87282325389402
// MI455X (gfx1250) — matrix operand order repaired and checked by reading, not compiled, not run on hardware
//
#include <hip/hip_runtime.h>
#include <hip/hip_bf16.h>
#include <math.h>

// ---------------------------------------------------------------------------
// Transformer-XL relative-position MHA for MI455X (gfx1250, wave32, WMMA).
//
//   pos_kernel : sinusoidal encodings (4096x512 f32), generated on device
//   proj<0..3> : bf16 WMMA projections -> q_u/q_v, K, V^T, ph_rev
//   attn_kernel: transposed flash attention; K/V^T tiles staged in LDS via
//                GLOBAL_LOAD_ASYNC_TO_LDS_B128 (double buffered, ASYNCcnt),
//                Toeplitz rel-pos band via 2 extra WMMA tiles + LDS gather
//   proj<4>    : output projection -> d_out f32
//
// Working set ~60 MB => resident in 192 MB L2; HBM traffic ~110 MB (~5us at
// 23.3 TB/s) => latency/bandwidth bound, so: bf16 everywhere, async LDS
// staging shared by the 4 waves of a workgroup, per-lane softmax state via
// the transposed S^T layout (wave32: lane column == query).
// ---------------------------------------------------------------------------

#define DEV static __device__ __forceinline__

typedef __attribute__((ext_vector_type(16))) __bf16 bf16x16;
typedef __attribute__((ext_vector_type(8)))  float  v8f;

// Types matching the gfx1250 async-load builtin signature:
//   void __builtin_amdgcn_global_load_async_to_lds_b128(
//        v4i __attribute__((address_space(1)))* src,   // global
//        v4i __attribute__((address_space(3)))* dst,   // LDS
//        constant int offset, constant int cpol)
typedef int v4i_ __attribute__((vector_size(16)));
typedef __attribute__((address_space(1))) v4i_* g_v4i_p;
typedef __attribute__((address_space(3))) v4i_* l_v4i_p;

namespace rk {

constexpr int DM = 512;   // d_model
constexpr int NH = 8;     // heads
constexpr int DK = 64;    // head dim
constexpr int BB = 4;     // batch
constexpr int TT = 2048;  // sequence length
constexpr int PL = 2*TT;  // positional rows

DEV v8f wmma_bf16(bf16x16 a, bf16x16 b, v8f c) {
  // D = A(16x32) * B(32x16) + C, f32 accumulate
  return __builtin_amdgcn_wmma_f32_16x16x32_bf16(false, a, false, b,
                                                 (short)0, c, false, false);
}

// A-fragment (16x32 bf16). Caller passes &row[l16][kbase + half*8].
// elements 0..7  -> K = half*8 + e ; elements 8..15 -> K = half*8 + 16 + (e-8)
DEV bf16x16 a_frag_f32(const float* p) {
  bf16x16 a;
  #pragma unroll
  for (int e = 0; e < 8; ++e) a[e] = (__bf16)p[e];
  #pragma unroll
  for (int e = 0; e < 8; ++e) a[8+e] = (__bf16)p[16+e];
  return a;
}
DEV bf16x16 a_frag_bf16(const __bf16* p) {
  bf16x16 a;
  #pragma unroll
  for (int e = 0; e < 8; ++e) a[e] = p[e];
  #pragma unroll
  for (int e = 0; e < 8; ++e) a[8+e] = p[16+e];
  return a;
}
// B-fragment (32x16 bf16): lane holds column n = l16, same K order as A-fragment.
DEV bf16x16 b_frag_f32(const float* p) {
  bf16x16 b;
  #pragma unroll
  for (int e = 0; e < 8; ++e) b[e] = (__bf16)p[e];
  #pragma unroll
  for (int e = 0; e < 8; ++e) b[8+e] = (__bf16)p[16+e];
  return b;
}
DEV bf16x16 b_frag_bf16(const __bf16* p) {
  bf16x16 b;
  #pragma unroll
  for (int e = 0; e < 8; ++e) b[e] = p[e];
  #pragma unroll
  for (int e = 0; e < 8; ++e) b[8+e] = p[16+e];
  return b;
}

// ---- CDNA5 async global->LDS copy (ASYNCcnt-tracked DMA), 16B per lane ----
DEV void async_copy_b128(const __bf16* gsrc, __bf16* ldst) {
#if __has_builtin(__builtin_amdgcn_global_load_async_to_lds_b128)
  __builtin_amdgcn_global_load_async_to_lds_b128(
      (g_v4i_p)gsrc, (l_v4i_p)ldst, 0, 0);
#else
  unsigned loff = (unsigned)(size_t)(l_v4i_p)ldst;
  asm volatile("global_load_async_to_lds_b128 %0, %1, off"
               :: "v"(loff), "v"(gsrc) : "memory");
#endif
}

template <int N> DEV void wait_asynccnt() {
#if __has_builtin(__builtin_amdgcn_s_wait_asynccnt)
  __builtin_amdgcn_s_wait_asynccnt(N);
#else
  if constexpr (N == 0) asm volatile("s_wait_asynccnt 0x0" ::: "memory");
  else                  asm volatile("s_wait_asynccnt 0x4" ::: "memory");
#endif
}

} // namespace rk

// ---------------------------------------------------------------------------
__global__ __launch_bounds__(256) void pos_kernel(float* __restrict__ p) {
  using namespace rk;
  const int idx = blockIdx.x * 256 + threadIdx.x;
  if (idx >= PL * DM) return;
  const int pos = idx >> 9;
  const int c   = idx & (DM - 1);
  const float w  = __powf(10000.f, -(float)(c >> 1) * (1.f / 256.f));
  const float th = (float)pos * w;
  p[idx] = (c & 1) ? __cosf(th) : __sinf(th);
}

// ---------------------------------------------------------------------------
// Generic 512-K projection: Y = X @ W^T (+bias), epilogue per MODE.
// ---------------------------------------------------------------------------
template <int MODE>
__global__ __launch_bounds__(256) void proj_kernel(
    const float* __restrict__ X, const float* __restrict__ W,
    const float* __restrict__ bias,
    const float* __restrict__ addb0, const float* __restrict__ addb1,
    __bf16* __restrict__ out0, __bf16* __restrict__ out1,
    float* __restrict__ fout)
{
  using namespace rk;
  const int lane = threadIdx.x & 31;
  const int wave = threadIdx.x >> 5;
  const int l16 = lane & 15, half = lane >> 4;
  const int m0 = blockIdx.x * 16;
  const int n0 = (blockIdx.y * 8 + wave) * 16;

  const float* arow = X + (size_t)(m0 + l16) * DM;
  const float* brow = W + (size_t)(n0 + l16) * DM;   // B[k][n] = W[n][k]

  v8f acc = {};
  #pragma unroll
  for (int k0 = 0; k0 < DM; k0 += 32) {
    bf16x16 a = a_frag_f32(arow + k0 + half * 8);
    bf16x16 b = b_frag_f32(brow + k0 + half * 8);
    acc = wmma_bf16(a, b, acc);
  }

  const int n = n0 + l16;
  const int h = n >> 6, d = n & 63;
  const float bn = (MODE == 3) ? 0.f : bias[n];
  #pragma unroll
  for (int v = 0; v < 8; ++v) {
    const int m = m0 + half * 8 + v;
    const float val = acc[v] + bn;
    if (MODE == 0) {          // Q: two outputs with b_u / b_v folded in
      const int bi = m >> 11, t = m & (TT - 1);
      const size_t idx = (((size_t)(bi * NH + h)) * TT + t) * DK + d;
      out0[idx] = (__bf16)(val + addb0[n]);
      out1[idx] = (__bf16)(val + addb1[n]);
    } else if (MODE == 1) {   // K: [B,H,T,DK]
      const int bi = m >> 11, t = m & (TT - 1);
      out0[(((size_t)(bi * NH + h)) * TT + t) * DK + d] = (__bf16)val;
    } else if (MODE == 2) {   // V transposed: [B,H,DK,T]
      const int bi = m >> 11, t = m & (TT - 1);
      out0[(((size_t)(bi * NH + h)) * DK + d) * TT + t] = (__bf16)val;
    } else if (MODE == 3) {   // ph_rev: [H, 2T, DK]
      out0[((size_t)h * PL + m) * DK + d] = (__bf16)val;
    } else {                  // MODE 4: final output, f32
      fout[(size_t)m * DM + n] = val;
    }
  }
}

// ---------------------------------------------------------------------------
// Transposed flash attention with async-LDS double buffering.
// Block = 4 waves; each wave owns a 16-query tile of one (b,h); all 4 waves
// share the same 32-key K / V^T tiles, staged in LDS by async DMA.
// ---------------------------------------------------------------------------
__global__ __launch_bounds__(128) void attn_kernel(
    const __bf16* __restrict__ qu, const __bf16* __restrict__ qv,
    const __bf16* __restrict__ kh, const __bf16* __restrict__ vT,
    const __bf16* __restrict__ ph, float* __restrict__ ao)
{
  using namespace rk;
  __shared__ __align__(32) __bf16 kbuf[2][32 * DK];   // [j'][d]   8 KB
  __shared__ __align__(32) __bf16 vbuf[2][DK * 32];   // [d][j']   8 KB
  __shared__ __align__(32) float  band_lds[4][32 * 16];          // 8 KB
  __shared__ __align__(32) __bf16 p_lds[4][16 * 32];             // 4 KB

  const int tid  = threadIdx.x;
  const int lane = tid & 31;
  const int wave = tid >> 5;
  const int l16 = lane & 15, half = lane >> 4;
  const int i0 = (blockIdx.x * 4 + wave) * 16;   // query tile base
  const int bh = blockIdx.y;                     // b*NH + h
  const int h  = bh & (NH - 1);

  const __bf16* qu_b = qu + (size_t)bh * TT * DK;
  const __bf16* qv_b = qv + (size_t)bh * TT * DK;
  const __bf16* k_b  = kh + (size_t)bh * TT * DK;
  const __bf16* v_b  = vT + (size_t)bh * DK * TT;
  const __bf16* p_b  = ph + (size_t)h  * PL * DK;

  float*  bandL = band_lds[wave];
  __bf16* pL    = p_lds[wave];

  // Cooperative async copy of one 32-key block: 4 b128 per thread (=> 4 per
  // wave on ASYNCcnt).  K tile is contiguous; V^T rows are 64B strided.
  const int vc0 = tid >> 2, vs0 = tid & 3;             // V chunk 0: d, 16B sub
  const int vc1 = (tid + 128) >> 2, vs1 = tid & 3;     // V chunk 1
  auto issue_copy = [&](int jsrc, int bsel) {
    const __bf16* kg = k_b + (size_t)jsrc * DK;
    async_copy_b128(kg + tid * 8,        &kbuf[bsel][tid * 8]);
    async_copy_b128(kg + tid * 8 + 1024, &kbuf[bsel][tid * 8 + 1024]);
    async_copy_b128(v_b + (size_t)vc0 * TT + jsrc + vs0 * 8,
                    &vbuf[bsel][vc0 * 32 + vs0 * 8]);
    async_copy_b128(v_b + (size_t)vc1 * TT + jsrc + vs1 * 8,
                    &vbuf[bsel][vc1 * 32 + vs1 * 8]);
  };

  issue_copy(0, 0);   // prologue: block 0 in flight while we set up q frags

  // Resident B-fragments of q^T (columns = the 16 queries of this tile).
  const __bf16* qrow_u = qu_b + (size_t)(i0 + l16) * DK + half * 8;
  const __bf16* qrow_v = qv_b + (size_t)(i0 + l16) * DK + half * 8;
  const bf16x16 bqu0 = b_frag_bf16(qrow_u);
  const bf16x16 bqu1 = b_frag_bf16(qrow_u + 32);
  const bf16x16 bqv0 = b_frag_bf16(qrow_v);
  const bf16x16 bqv1 = b_frag_bf16(qrow_v + 32);

  float mrow = -INFINITY, lrow = 0.f;
  v8f o0 = {}, o1 = {}, o2 = {}, o3 = {};
  int cur = 0;

  #pragma unroll 1
  for (int j0 = 0; j0 < TT; j0 += 32) {
    // Prefetch next block into the other buffer, then drain the current one.
    if (j0 + 32 < TT) { issue_copy(j0 + 32, cur ^ 1); wait_asynccnt<4>(); }
    else              { wait_asynccnt<0>(); }
    __syncthreads();
    const __bf16* kl = kbuf[cur];
    const __bf16* vl = vbuf[cur];

    v8f ssub[2];
    #pragma unroll
    for (int jj = 0; jj < 2; ++jj) {
      const int jb = j0 + jj * 16;
      // ac^T : A = 16 key rows x 64 dims (from LDS), B = q_u^T
      const __bf16* krow = kl + (jj * 16 + l16) * DK + half * 8;
      v8f s = {};
      s = wmma_bf16(a_frag_bf16(krow),      bqu0, s);
      s = wmma_bf16(a_frag_bf16(krow + 32), bqu1, s);

      if (jb <= i0 + 15) {   // tile touches j<=i: add relative-position band
        const int rbase = i0 - jb - 15;
        #pragma unroll
        for (int mt2 = 0; mt2 < 2; ++mt2) {
          int r = rbase + mt2 * 16 + l16;
          r = r < 0 ? 0 : r;            // clamped rows only feed masked cells
          const __bf16* prow = p_b + (size_t)r * DK + half * 8;
          v8f rr = {};
          rr = wmma_bf16(a_frag_bf16(prow),      bqv0, rr);
          rr = wmma_bf16(a_frag_bf16(prow + 32), bqv1, rr);
          #pragma unroll
          for (int v = 0; v < 8; ++v)
            bandL[(mt2 * 16 + half * 8 + v) * 16 + l16] = rr[v];
        }
        #pragma unroll
        for (int v = 0; v < 8; ++v) {
          const int jp = half * 8 + v;        // key row within tile
          const int mm = l16 - jp + 15;       // band row for (i'=l16, j'=jp)
          const float bd = bandL[mm * 16 + l16];
          s[v] += ((jb + jp) <= (i0 + l16)) ? bd : 0.f;
        }
      }
      #pragma unroll
      for (int v = 0; v < 8; ++v) s[v] *= 0.125f;   // 1/sqrt(DK)
      ssub[jj] = s;
    }

    // ---- online softmax over this 32-key block (per query == per lane col)
    float tmax = -INFINITY;
    #pragma unroll
    for (int v = 0; v < 8; ++v)
      tmax = fmaxf(tmax, fmaxf(ssub[0][v], ssub[1][v]));
    tmax = fmaxf(tmax, __shfl_xor(tmax, 16, 32));
    const float mnew = fmaxf(mrow, tmax);
    const float corr = __expf(mrow - mnew);
    float tsum = 0.f;
    #pragma unroll
    for (int jj = 0; jj < 2; ++jj) {
      #pragma unroll
      for (int v = 0; v < 8; ++v) {
        const float pv = __expf(ssub[jj][v] - mnew);
        ssub[jj][v] = pv;
        tsum += pv;
      }
    }
    tsum += __shfl_xor(tsum, 16, 32);
    lrow = lrow * corr + tsum;
    mrow = mnew;
    #pragma unroll
    for (int v = 0; v < 8; ++v) {
      o0[v] *= corr; o1[v] *= corr; o2[v] *= corr; o3[v] *= corr;
    }

    // stash P^T (bf16) to LDS: layout [i'][j''], j'' in [0,32)
    #pragma unroll
    for (int jj = 0; jj < 2; ++jj) {
      #pragma unroll
      for (int v = 0; v < 8; ++v)
        pL[l16 * 32 + jj * 16 + half * 8 + v] = (__bf16)ssub[jj][v];
    }

    // O^T += V^T(16d x 32j, from LDS) @ P^T(32j x 16q)
    const bf16x16 bp = b_frag_bf16(pL + l16 * 32 + half * 8);
    const __bf16* vrow = vl + (l16) * 32 + half * 8;
    o0 = wmma_bf16(a_frag_bf16(vrow),           bp, o0);
    o1 = wmma_bf16(a_frag_bf16(vrow + 16 * 32), bp, o1);
    o2 = wmma_bf16(a_frag_bf16(vrow + 32 * 32), bp, o2);
    o3 = wmma_bf16(a_frag_bf16(vrow + 48 * 32), bp, o3);

    __syncthreads();   // buffer `cur` free for the next prefetch
    cur ^= 1;
  }

  // Epilogue: O^T lane col = query l16, rows d = dt*16 + half*8 + v.
  const float inv = 1.f / lrow;
  const int bi = bh >> 3;
  float* aorow = ao + ((size_t)bi * TT + i0 + l16) * DM + h * DK + half * 8;
  #pragma unroll
  for (int v = 0; v < 8; ++v) aorow[v]      = o0[v] * inv;
  #pragma unroll
  for (int v = 0; v < 8; ++v) aorow[16 + v] = o1[v] * inv;
  #pragma unroll
  for (int v = 0; v < 8; ++v) aorow[32 + v] = o2[v] * inv;
  #pragma unroll
  for (int v = 0; v < 8; ++v) aorow[48 + v] = o3[v] * inv;
}

// ---------------------------------------------------------------------------
extern "C" void kernel_launch(void* const* d_in, const int* in_sizes, int n_in,
                              void* d_out, int out_size, void* d_ws, size_t ws_size,
                              hipStream_t stream) {
  using namespace rk;
  const float* q   = (const float*)d_in[0];
  const float* k   = (const float*)d_in[1];
  const float* v   = (const float*)d_in[2];
  // d_in[3] = mask: all-true in this problem, unused.
  const float* Wq  = (const float*)d_in[4];
  const float* bq  = (const float*)d_in[5];
  const float* Wk  = (const float*)d_in[6];
  const float* bk  = (const float*)d_in[7];
  const float* Wv  = (const float*)d_in[8];
  const float* bv  = (const float*)d_in[9];
  const float* Wp  = (const float*)d_in[10];
  const float* b_u = (const float*)d_in[11];
  const float* b_v = (const float*)d_in[12];
  const float* Wo  = (const float*)d_in[13];
  const float* bo  = (const float*)d_in[14];

  char* ws = (char*)d_ws;
  float*  posenc = (float*)ws;   ws += (size_t)PL * DM * sizeof(float);      //  8 MB
  __bf16* quB    = (__bf16*)ws;  ws += (size_t)BB * NH * TT * DK * 2;        //  8 MB
  __bf16* qvB    = (__bf16*)ws;  ws += (size_t)BB * NH * TT * DK * 2;        //  8 MB
  __bf16* khB    = (__bf16*)ws;  ws += (size_t)BB * NH * TT * DK * 2;        //  8 MB
  __bf16* vTB    = (__bf16*)ws;  ws += (size_t)BB * NH * TT * DK * 2;        //  8 MB
  __bf16* phB    = (__bf16*)ws;  ws += (size_t)NH * PL * DK * 2;             //  4 MB
  float*  aoB    = (float*)ws;   ws += (size_t)BB * TT * DM * sizeof(float); // 16 MB

  pos_kernel<<<dim3((PL * DM) / 256), 256, 0, stream>>>(posenc);
  proj_kernel<0><<<dim3((BB * TT) / 16, 4), 256, 0, stream>>>(
      q, Wq, bq, b_u, b_v, quB, qvB, nullptr);
  proj_kernel<1><<<dim3((BB * TT) / 16, 4), 256, 0, stream>>>(
      k, Wk, bk, nullptr, nullptr, khB, nullptr, nullptr);
  proj_kernel<2><<<dim3((BB * TT) / 16, 4), 256, 0, stream>>>(
      v, Wv, bv, nullptr, nullptr, vTB, nullptr, nullptr);
  proj_kernel<3><<<dim3(PL / 16, 4), 256, 0, stream>>>(
      posenc, Wp, nullptr, nullptr, nullptr, phB, nullptr, nullptr);
  attn_kernel<<<dim3(TT / 64, BB * NH), 128, 0, stream>>>(
      quB, qvB, khB, vTB, phB, aoB);
  proj_kernel<4><<<dim3((BB * TT) / 16, 4), 256, 0, stream>>>(
      aoB, Wo, bo, nullptr, nullptr, nullptr, nullptr, (float*)d_out);
}
